// MPNNClassifier_77790447665502
// MI455X (gfx1250) — matrix operand order repaired and checked by reading, not compiled, not run on hardware
//
#include <hip/hip_runtime.h>
#include <hip/hip_bf16.h>
#include <math.h>

// ---------------------------------------------------------------------------
// MPNN classifier forward on gfx1250 (MI455X), f16 WMMA with f32 accumulate.
// ---------------------------------------------------------------------------

typedef _Float16 v16h __attribute__((ext_vector_type(16)));
typedef _Float16 v8h  __attribute__((ext_vector_type(8)));
typedef float    v8f  __attribute__((ext_vector_type(8)));

#define N_NODES  100000
#define N_EDGES  800000
#define HID      64
#define N_GRAPHS 256

union V16 { v16h v; v8h h[2]; };

__device__ __forceinline__ v8f wmma_f16(v16h a, v16h b, v8f c) {
  // D = A(16x32 f16) * B(32x16 f16) + C(16x16 f32)
  return __builtin_amdgcn_wmma_f32_16x16x32_f16(false, a, false, b,
                                                (short)0, c, false, false);
}

__device__ __forceinline__ void atomic_add_f32(float* p, float v) {
  __hip_atomic_fetch_add(p, v, __ATOMIC_RELAXED, __HIP_MEMORY_SCOPE_AGENT);
}

__device__ __forceinline__ v8h cvt8(float4 a, float4 b) {
  v8h r;
  r[0] = (_Float16)a.x; r[1] = (_Float16)a.y; r[2] = (_Float16)a.z; r[3] = (_Float16)a.w;
  r[4] = (_Float16)b.x; r[5] = (_Float16)b.y; r[6] = (_Float16)b.z; r[7] = (_Float16)b.w;
  return r;
}

// ---------------------------------------------------------------------------
// Kernel 0: h0 = node_emb[x] (f32 + f16 copies), zero message accum + output.
// ---------------------------------------------------------------------------
__global__ __launch_bounds__(256) void k0_init(
    const int* __restrict__ x, const float* __restrict__ node_emb,
    float* __restrict__ h0f32, _Float16* __restrict__ h0f16,
    float* __restrict__ mbuf, float* __restrict__ out)
{
  long i = (long)blockIdx.x * blockDim.x + threadIdx.x;
  const long total = (long)N_NODES * HID;
  if (i < total) {
    int node = (int)(i >> 6);
    int f = (int)(i & 63);
    float v = node_emb[(long)x[node] * HID + f];
    h0f32[i] = v;
    h0f16[i] = (_Float16)v;
    mbuf[i] = 0.0f;
  }
  if (i < N_GRAPHS) out[i] = 0.0f;
}

// ---------------------------------------------------------------------------
// Kernel 1: edge MLP + scatter-add.  msg = relu([h0[src],e]@W1+b1)@W2+b2
// One wave handles a tile of 16 edges.  B operands pre-shuffled into per-lane
// WMMA fragment layout in LDS once per workgroup.
// ---------------------------------------------------------------------------
#define K1_WAVES 4
__global__ __launch_bounds__(K1_WAVES * 32) void k1_edge(
    const int* __restrict__ src, const int* __restrict__ dst,
    const int* __restrict__ attr, const _Float16* __restrict__ h0f16,
    const float* __restrict__ edge_emb,
    const float* __restrict__ W1, const float* __restrict__ b1,
    const float* __restrict__ W2, const float* __restrict__ b2,
    float* __restrict__ mbuf)
{
  __shared__ __align__(16) _Float16 w1f[16 * 32 * 16];          // 16 KB: 4 Ntiles x 4 Kchunks
  __shared__ __align__(16) _Float16 w2f[8 * 32 * 16];           //  8 KB: 4 Ntiles x 2 Kchunks
  __shared__ float b1s[64], b2s[64];
  __shared__ __align__(16) _Float16 eemb[64][72];               //  9 KB (padded)
  __shared__ __align__(16) _Float16 cat[K1_WAVES][16][136];     // 17 KB (padded, per wave)
  __shared__ __align__(16) _Float16 hid[K1_WAVES][16][72];      //  9 KB (padded, per wave)

  const int tid = threadIdx.x;
  // Build B fragments: frag element j of lane l <-> K = 32c + 8*(l/16) + (j&7) + 16*(j>>3), N = n*16 + (l&15)
  for (int i = tid; i < 16 * 32 * 16; i += blockDim.x) {
    int j = i & 15, l = (i >> 4) & 31, nc = i >> 9;
    int c = nc & 3, n = nc >> 2;
    int k = 32 * c + 8 * (l >> 4) + (j & 7) + ((j >> 3) << 4);
    int col = n * 16 + (l & 15);
    w1f[i] = (_Float16)W1[k * 64 + col];
  }
  for (int i = tid; i < 8 * 32 * 16; i += blockDim.x) {
    int j = i & 15, l = (i >> 4) & 31, nc = i >> 9;
    int c = nc & 1, n = nc >> 1;
    int k = 32 * c + 8 * (l >> 4) + (j & 7) + ((j >> 3) << 4);
    int col = n * 16 + (l & 15);
    w2f[i] = (_Float16)W2[k * 64 + col];
  }
  for (int i = tid; i < 64; i += blockDim.x) { b1s[i] = b1[i]; b2s[i] = b2[i]; }
  for (int i = tid; i < 64 * 64; i += blockDim.x)
    eemb[i >> 6][i & 63] = (_Float16)edge_emb[i];
  __syncthreads();

  const int wave = tid >> 5, lane = tid & 31;
  const int row = lane & 15;            // A-frag row (edge within tile)
  const int kb  = (lane >> 4) * 8;      // A-frag K sub-offset per half-wave
  const int col = lane & 15;            // C/D column within N-tile
  const int nTiles = N_EDGES / 16;      // 50000, exact

  for (int tile = blockIdx.x * K1_WAVES + wave; tile < nTiles;
       tile += gridDim.x * K1_WAVES) {
    const int e0 = tile * 16;
    int srcv = 0, dstv = 0, attrv = 0;
    if (lane < 16) { srcv = src[e0 + lane]; dstv = dst[e0 + lane]; attrv = attr[e0 + lane]; }

    // Stage cat = [h0[src] | edge_emb[attr]] as f16, 16 rows x 128 cols.
#pragma unroll
    for (int j = 0; j < 2; ++j) {
      int chunk = lane + 32 * j;        // 0..63
      int r = chunk >> 2, part = chunk & 3;
      int s = __shfl(srcv, r, 32);
      int a = __shfl(attrv, r, 32);
      *(v8h*)&cat[wave][r][part * 8]      = *(const v8h*)&h0f16[(long)s * 64 + part * 8];
      *(v8h*)&cat[wave][r][64 + part * 8] = *(const v8h*)&eemb[a][part * 8];
    }
    __builtin_amdgcn_fence(__ATOMIC_ACQ_REL, "workgroup");

    // A fragments: lane holds row, K = {32c+kb..+7} and {32c+kb+16..+23}
    V16 af[4];
#pragma unroll
    for (int c = 0; c < 4; ++c) {
      af[c].h[0] = *(const v8h*)&cat[wave][row][c * 32 + kb];
      af[c].h[1] = *(const v8h*)&cat[wave][row][c * 32 + kb + 16];
    }

    // Layer 1: [16x128] @ [128x64] + b1, ReLU -> hid (f16, C layout -> row major LDS)
#pragma unroll
    for (int n = 0; n < 4; ++n) {
      float bb = b1s[n * 16 + col];
      v8f acc = {bb, bb, bb, bb, bb, bb, bb, bb};
#pragma unroll
      for (int c = 0; c < 4; ++c) {
        v16h bf = *(const v16h*)&w1f[((n * 4 + c) * 32 + lane) * 16];
        acc = wmma_f16(af[c].v, bf, acc);
      }
#pragma unroll
      for (int r = 0; r < 8; ++r) {
        float vv = acc[r];
        vv = vv > 0.f ? vv : 0.f;
        hid[wave][r + 8 * (lane >> 4)][n * 16 + col] = (_Float16)vv;
      }
    }
    __builtin_amdgcn_fence(__ATOMIC_ACQ_REL, "workgroup");

    V16 hf[2];
#pragma unroll
    for (int c = 0; c < 2; ++c) {
      hf[c].h[0] = *(const v8h*)&hid[wave][row][c * 32 + kb];
      hf[c].h[1] = *(const v8h*)&hid[wave][row][c * 32 + kb + 16];
    }

    // Layer 2: [16x64] @ [64x64] + b2, scatter-add into m[dst]
#pragma unroll
    for (int n = 0; n < 4; ++n) {
      float bb = b2s[n * 16 + col];
      v8f acc = {bb, bb, bb, bb, bb, bb, bb, bb};
#pragma unroll
      for (int c = 0; c < 2; ++c) {
        v16h bf = *(const v16h*)&w2f[((n * 2 + c) * 32 + lane) * 16];
        acc = wmma_f16(hf[c].v, bf, acc);
      }
#pragma unroll
      for (int r = 0; r < 8; ++r) {
        int M = r + 8 * (lane >> 4);
        int nd = __shfl(dstv, M, 32);
        atomic_add_f32(&mbuf[(long)nd * 64 + n * 16 + col], acc[r]);
      }
    }
  }
}

// ---------------------------------------------------------------------------
// Kernel 2: GRU cell.  gx = m@Wih+bih, gh = h0@Whh+bhh (both in WMMA C layout
// so the gate math is pure per-lane VALU).  hT stored as f16.
// ---------------------------------------------------------------------------
__global__ __launch_bounds__(256) void k2_gru(
    const float* __restrict__ mbuf, const _Float16* __restrict__ h0f16,
    const float* __restrict__ h0f32,
    const float* __restrict__ Wih, const float* __restrict__ bih,
    const float* __restrict__ Whh, const float* __restrict__ bhh,
    _Float16* __restrict__ hT)
{
  __shared__ __align__(16) _Float16 wihf[24 * 32 * 16];   // 24 KB: 3 gates x 4 Ntiles x 2 Kchunks
  __shared__ __align__(16) _Float16 whhf[24 * 32 * 16];   // 24 KB
  __shared__ float bihs[192], bhhs[192];

  const int tid = threadIdx.x;
  for (int i = tid; i < 24 * 32 * 16; i += blockDim.x) {
    int j = i & 15, l = (i >> 4) & 31, f = i >> 9;
    int c = f & 1, t = (f >> 1) & 3, g = f >> 3;
    int k = 32 * c + 8 * (l >> 4) + (j & 7) + ((j >> 3) << 4);
    int colG = g * 64 + t * 16 + (l & 15);
    wihf[i] = (_Float16)Wih[k * 192 + colG];
    whhf[i] = (_Float16)Whh[k * 192 + colG];
  }
  for (int i = tid; i < 192; i += blockDim.x) { bihs[i] = bih[i]; bhhs[i] = bhh[i]; }
  __syncthreads();

  const int wave = tid >> 5, lane = tid & 31;
  const int row = lane & 15, kb = (lane >> 4) * 8, col = lane & 15;
  const int wpb = blockDim.x >> 5;
  const int nTiles = N_NODES / 16;     // 6250, exact

  for (int tile = blockIdx.x * wpb + wave; tile < nTiles; tile += gridDim.x * wpb) {
    const int node0 = tile * 16;
    const long base = (long)(node0 + row) * 64;

    V16 am[2], ah[2];
#pragma unroll
    for (int c = 0; c < 2; ++c) {
      const float* p = mbuf + base + c * 32 + kb;
      am[c].h[0] = cvt8(*(const float4*)p,        *(const float4*)(p + 4));
      am[c].h[1] = cvt8(*(const float4*)(p + 16), *(const float4*)(p + 20));
      const _Float16* q = h0f16 + base + c * 32 + kb;
      ah[c].h[0] = *(const v8h*)q;
      ah[c].h[1] = *(const v8h*)(q + 16);
    }

    v8f rg[4], zg[4];
#pragma unroll
    for (int g = 0; g < 2; ++g) {       // g=0: reset gate r, g=1: update gate z
#pragma unroll
      for (int t = 0; t < 4; ++t) {
        int cg = g * 64 + t * 16 + col;
        float bx = bihs[cg], bh = bhhs[cg];
        v8f ax = {bx, bx, bx, bx, bx, bx, bx, bx};
        v8f hx = {bh, bh, bh, bh, bh, bh, bh, bh};
#pragma unroll
        for (int c = 0; c < 2; ++c) {
          int f = (g * 4 + t) * 2 + c;
          ax = wmma_f16(am[c].v, *(const v16h*)&wihf[(f * 32 + lane) * 16], ax);
          hx = wmma_f16(ah[c].v, *(const v16h*)&whhf[(f * 32 + lane) * 16], hx);
        }
        v8f o;
#pragma unroll
        for (int r = 0; r < 8; ++r) o[r] = 1.f / (1.f + expf(-(ax[r] + hx[r])));
        if (g == 0) rg[t] = o; else zg[t] = o;
      }
    }

#pragma unroll
    for (int t = 0; t < 4; ++t) {       // candidate gate n + blend
      int cg = 128 + t * 16 + col;
      float bx = bihs[cg], bh = bhhs[cg];
      v8f ax = {bx, bx, bx, bx, bx, bx, bx, bx};
      v8f hx = {bh, bh, bh, bh, bh, bh, bh, bh};
#pragma unroll
      for (int c = 0; c < 2; ++c) {
        int f = (8 + t) * 2 + c;
        ax = wmma_f16(am[c].v, *(const v16h*)&wihf[(f * 32 + lane) * 16], ax);
        hx = wmma_f16(ah[c].v, *(const v16h*)&whhf[(f * 32 + lane) * 16], hx);
      }
#pragma unroll
      for (int r = 0; r < 8; ++r) {
        int M = r + 8 * (lane >> 4);
        long idx = (long)(node0 + M) * 64 + t * 16 + col;
        float nv = tanhf(ax[r] + rg[t][r] * hx[r]);
        float z = zg[t][r];
        float ht = (1.f - z) * nv + z * h0f32[idx];
        hT[idx] = (_Float16)ht;
      }
    }
  }
}

// ---------------------------------------------------------------------------
// Kernel 3: readout MLPs + sigmoid + global_add_pool.
// ---------------------------------------------------------------------------
__global__ __launch_bounds__(256) void k3_readout(
    const _Float16* __restrict__ hT, const _Float16* __restrict__ h0f16,
    const int* __restrict__ batch,
    const float* __restrict__ riW1, const float* __restrict__ rib1,
    const float* __restrict__ riW2, const float* __restrict__ rib2,
    const float* __restrict__ rjW1, const float* __restrict__ rjb1,
    const float* __restrict__ rjW2, const float* __restrict__ rjb2,
    float* __restrict__ out)
{
  __shared__ __align__(16) _Float16 rif[16 * 32 * 16];    // 16 KB: 8 Ntiles x 2 Kchunks
  __shared__ __align__(16) _Float16 rjf[16 * 32 * 16];    // 16 KB
  __shared__ float rib1s[128], rjb1s[128], riw2s[128], rjw2s[128];

  const int tid = threadIdx.x;
  for (int i = tid; i < 16 * 32 * 16; i += blockDim.x) {
    int j = i & 15, l = (i >> 4) & 31, f = i >> 9;
    int c = f & 1, t = f >> 1;
    int k = 32 * c + 8 * (l >> 4) + (j & 7) + ((j >> 3) << 4);
    int colG = t * 16 + (l & 15);
    rif[i] = (_Float16)riW1[k * 128 + colG];
    rjf[i] = (_Float16)rjW1[k * 128 + colG];
  }
  for (int i = tid; i < 128; i += blockDim.x) {
    rib1s[i] = rib1[i]; rjb1s[i] = rjb1[i];
    riw2s[i] = riW2[i]; rjw2s[i] = rjW2[i];
  }
  __syncthreads();

  const float bi2 = rib2[0], bj2 = rjb2[0];
  const int wave = tid >> 5, lane = tid & 31;
  const int row = lane & 15, kb = (lane >> 4) * 8, col = lane & 15;
  const int wpb = blockDim.x >> 5;
  const int nTiles = N_NODES / 16;

  for (int tile = blockIdx.x * wpb + wave; tile < nTiles; tile += gridDim.x * wpb) {
    const int node0 = tile * 16;
    const long base = (long)(node0 + row) * 64;

    V16 aT[2], a0[2];
#pragma unroll
    for (int c = 0; c < 2; ++c) {
      const _Float16* p = hT + base + c * 32 + kb;
      aT[c].h[0] = *(const v8h*)p;        aT[c].h[1] = *(const v8h*)(p + 16);
      const _Float16* q = h0f16 + base + c * 32 + kb;
      a0[c].h[0] = *(const v8h*)q;        a0[c].h[1] = *(const v8h*)(q + 16);
    }

    float pi[8] = {0, 0, 0, 0, 0, 0, 0, 0};
    float pj[8] = {0, 0, 0, 0, 0, 0, 0, 0};
#pragma unroll
    for (int t = 0; t < 8; ++t) {
      float bb = rib1s[t * 16 + col];
      v8f ai = {bb, bb, bb, bb, bb, bb, bb, bb};
      bb = rjb1s[t * 16 + col];
      v8f aj = {bb, bb, bb, bb, bb, bb, bb, bb};
#pragma unroll
      for (int c = 0; c < 2; ++c) {
        int f = t * 2 + c;
        ai = wmma_f16(aT[c].v, *(const v16h*)&rif[(f * 32 + lane) * 16], ai);
        aj = wmma_f16(a0[c].v, *(const v16h*)&rjf[(f * 32 + lane) * 16], aj);
      }
      float wi = riw2s[t * 16 + col], wj = rjw2s[t * 16 + col];
#pragma unroll
      for (int r = 0; r < 8; ++r) {
        float vi = ai[r] > 0.f ? ai[r] : 0.f;
        float vj = aj[r] > 0.f ? aj[r] : 0.f;
        pi[r] += vi * wi;
        pj[r] += vj * wj;
      }
    }
    // butterfly reduce over the 16 lanes that share the same node rows
#pragma unroll
    for (int mask = 1; mask < 16; mask <<= 1) {
#pragma unroll
      for (int r = 0; r < 8; ++r) {
        pi[r] += __shfl_xor(pi[r], mask, 32);
        pj[r] += __shfl_xor(pj[r], mask, 32);
      }
    }
    if ((lane & 15) == 0) {
#pragma unroll
      for (int r = 0; r < 8; ++r) {
        int node = node0 + r + 8 * (lane >> 4);
        float s = pi[r] + bi2 + pj[r] + bj2;
        float prob = 1.f / (1.f + expf(-s));
        atomic_add_f32(&out[batch[node]], prob);
      }
    }
  }
}

// ---------------------------------------------------------------------------
extern "C" void kernel_launch(void* const* d_in, const int* in_sizes, int n_in,
                              void* d_out, int out_size, void* d_ws, size_t ws_size,
                              hipStream_t stream) {
  const int*   x        = (const int*)d_in[0];
  const int*   eidx     = (const int*)d_in[1];   // [2, E] row-major: src then dst
  const int*   eattr    = (const int*)d_in[2];
  const int*   batch    = (const int*)d_in[3];
  const float* node_emb = (const float*)d_in[5];
  const float* edge_emb = (const float*)d_in[6];
  const float* msg_W1   = (const float*)d_in[7];
  const float* msg_b1   = (const float*)d_in[8];
  const float* msg_W2   = (const float*)d_in[9];
  const float* msg_b2   = (const float*)d_in[10];
  const float* gru_Wih  = (const float*)d_in[11];
  const float* gru_bih  = (const float*)d_in[12];
  const float* gru_Whh  = (const float*)d_in[13];
  const float* gru_bhh  = (const float*)d_in[14];
  const float* ri_W1    = (const float*)d_in[15];
  const float* ri_b1    = (const float*)d_in[16];
  const float* ri_W2    = (const float*)d_in[17];
  const float* ri_b2    = (const float*)d_in[18];
  const float* rj_W1    = (const float*)d_in[19];
  const float* rj_b1    = (const float*)d_in[20];
  const float* rj_W2    = (const float*)d_in[21];
  const float* rj_b2    = (const float*)d_in[22];
  float* out = (float*)d_out;

  // Workspace layout (all 16B aligned): h0 f32 | h0 f16 | m f32 | hT f16
  char* ws = (char*)d_ws;
  float*     h0f32 = (float*)ws;                          // 25,600,000 B
  _Float16*  h0f16 = (_Float16*)(ws + 25600000);          // 12,800,000 B
  float*     mbuf  = (float*)(ws + 38400000);             // 25,600,000 B
  _Float16*  hTbuf = (_Float16*)(ws + 64000000);          // 12,800,000 B

  const long elems = (long)N_NODES * HID;                 // 6.4M
  k0_init<<<(int)((elems + 255) / 256), 256, 0, stream>>>(
      x, node_emb, h0f32, h0f16, mbuf, out);

  // 50000 edge tiles, 4 waves/block -> 12500 blocks, one tile per wave.
  k1_edge<<<12500, K1_WAVES * 32, 0, stream>>>(
      eidx, eidx + N_EDGES, eattr, h0f16, edge_emb,
      msg_W1, msg_b1, msg_W2, msg_b2, mbuf);

  // 6250 node tiles, 8 waves/block.
  k2_gru<<<(6250 + 7) / 8, 256, 0, stream>>>(
      mbuf, h0f16, h0f32, gru_Wih, gru_bih, gru_Whh, gru_bhh, hTbuf);

  k3_readout<<<(6250 + 7) / 8, 256, 0, stream>>>(
      hTbuf, h0f16, batch, ri_W1, ri_b1, ri_W2, ri_b2,
      rj_W1, rj_b1, rj_W2, rj_b2, out);
}
